// IEGMN_Layer_51393578664430
// MI455X (gfx1250) — matrix operand order repaired and checked by reading, not compiled, not run on hardware
//
#include <hip/hip_runtime.h>
#include <hip/hip_bf16.h>

typedef __attribute__((ext_vector_type(16))) _Float16 v16h;
typedef __attribute__((ext_vector_type(8)))  float    v8f;

union Frag16 { v16h v; _Float16 h[16]; };

__device__ __forceinline__ float lrelu(float x) { return x >= 0.f ? x : 0.01f * x; }

// A-fragment (16x32 f16) K index for half 'h' in a lane: one contiguous 16-half run.
__device__ __forceinline__ int amapK(int h, int half16) {
    return (half16 << 4) + h;
}

__device__ __forceinline__ v8f wmma16(const Frag16& a, const Frag16& b, v8f c) {
    return __builtin_amdgcn_wmma_f32_16x16x32_f16(false, a.v, false, b.v, (short)0, c, false, false);
}

__device__ __forceinline__ float redsum16(float x) {
    x += __shfl_xor(x, 1, 16);
    x += __shfl_xor(x, 2, 16);
    x += __shfl_xor(x, 4, 16);
    x += __shfl_xor(x, 8, 16);
    return x;
}

// ---------------------------------------------------------------------------
// Weight prep: transpose to N-major and convert to f16 (zero-padded K).
// ---------------------------------------------------------------------------
__global__ void prep_weights(const float* __restrict__ w1,   // [175][64]
                             const float* __restrict__ w2,   // [64][64]
                             const float* __restrict__ cw1,  // [64][64]
                             const float* __restrict__ nw1,  // [256][64]
                             const float* __restrict__ nw2,  // [64][64]
                             _Float16* __restrict__ w1t,     // [64][192]
                             _Float16* __restrict__ w2t,     // [64][64]
                             _Float16* __restrict__ cw1t,    // [64][64]
                             _Float16* __restrict__ nw1t,    // [64][256]
                             _Float16* __restrict__ nw2t)    // [64][64]
{
    int t = blockIdx.x * blockDim.x + threadIdx.x;
    int stride = gridDim.x * blockDim.x;
    for (int idx = t; idx < 64 * 192; idx += stride) {
        int n = idx / 192, k = idx % 192;
        w1t[idx] = (k < 175) ? (_Float16)w1[k * 64 + n] : (_Float16)0.f;
    }
    for (int idx = t; idx < 64 * 64; idx += stride) {
        int n = idx / 64, k = idx % 64;
        w2t[idx]  = (_Float16)w2[k * 64 + n];
        cw1t[idx] = (_Float16)cw1[k * 64 + n];
        nw2t[idx] = (_Float16)nw2[k * 64 + n];
    }
    for (int idx = t; idx < 64 * 256; idx += stride) {
        int n = idx / 256, k = idx % 256;
        nw1t[idx] = (_Float16)nw1[k * 64 + n];
    }
}

// ---------------------------------------------------------------------------
// q = lrelu(h@Wq), k = lrelu(h@Wk), v = h@Wv + b.  v stored transposed [64][N].
// ---------------------------------------------------------------------------
__global__ void qkv_kernel(const float* __restrict__ h,
                           const float* __restrict__ qw, const float* __restrict__ kw,
                           const float* __restrict__ vw, const float* __restrict__ vb,
                           _Float16* __restrict__ qh, _Float16* __restrict__ kh,
                           _Float16* __restrict__ vt, int n)
{
    int t = blockIdx.x * blockDim.x + threadIdx.x;
    if (t >= n * 64) return;
    int i = t >> 6, j = t & 63;
    const float* hr = h + (size_t)i * 64;
    float aq = 0.f, ak = 0.f, av = 0.f;
    #pragma unroll 8
    for (int k = 0; k < 64; ++k) {
        float hv = hr[k];
        aq = fmaf(hv, qw[k * 64 + j], aq);
        ak = fmaf(hv, kw[k * 64 + j], ak);
        av = fmaf(hv, vw[k * 64 + j], av);
    }
    qh[t] = (_Float16)lrelu(aq);
    kh[t] = (_Float16)lrelu(ak);
    vt[(size_t)j * n + i] = (_Float16)(av + vb[j]);
}

// ---------------------------------------------------------------------------
// Flash attention, both sides in one launch (blockIdx.y selects side).
// One wave per 16-row tile; 32 columns per iteration; block-uniform running
// max (wave-wide) so P = exp(a - M) <= 1 is f16-safe; softmax denominator
// via an extra WMMA with an all-ones B fragment (P @ ones).
// LDC != 0 specializes the leading dimension (== ncols == mask_ld for both
// sides) at compile time so strides fold into 24-bit instruction offsets.
// Mask is streamed with non-temporal loads (read-once 256MB; keep K/V in L2).
// side 0: mv = mask[row*ld + col]; side 1: mv = mask[col*ld + row]
// ---------------------------------------------------------------------------
template <int LDC>
__global__ __launch_bounds__(256) void attn_kernel(
    const _Float16* __restrict__ Q0, const _Float16* __restrict__ K0,
    const _Float16* __restrict__ V0, float* __restrict__ out0,
    const _Float16* __restrict__ Q1, const _Float16* __restrict__ K1,
    const _Float16* __restrict__ V1, float* __restrict__ out1,
    const float* __restrict__ mask,
    int nrows0, int ncols0, int nrows1, int ncols1, int mask_ld_rt)
{
    const int side = blockIdx.y;
    const _Float16* __restrict__ Q  = side ? Q1 : Q0;
    const _Float16* __restrict__ Km = side ? K1 : K0;
    const _Float16* __restrict__ Vt = side ? V1 : V0;
    float* __restrict__ out = side ? out1 : out0;
    const int nrows = side ? nrows1 : nrows0;
    const int ncols   = LDC ? LDC : (side ? ncols1 : ncols0);
    const int mask_ld = LDC ? LDC : mask_ld_rt;

    __shared__ _Float16 sP[8][16 * 32];
    int wave = threadIdx.x >> 5;
    int lane = threadIdx.x & 31;
    int half16 = lane >> 4;
    int ln = lane & 15;
    int tile = blockIdx.x * 8 + wave;
    if (tile * 16 >= nrows) return;

    int m = tile * 16 + ln;
    Frag16 aq0, aq1;
    #pragma unroll
    for (int h = 0; h < 16; ++h) {
        int k = amapK(h, half16);
        aq0.h[h] = Q[(size_t)m * 64 + k];
        aq1.h[h] = Q[(size_t)m * 64 + 32 + k];
    }
    Frag16 bones;
    #pragma unroll
    for (int h = 0; h < 16; ++h) bones.h[h] = (_Float16)1.f;

    // loop-invariant base pointers (advanced by constant strides per iter)
    const _Float16* kbase = Km + (size_t)ln * 64 + half16 * 16;
    const _Float16* vbase = Vt + (size_t)ln * ncols + half16 * 16;
    const float* mbase0;   // side 0: row base (col advances); side 1: col base
    if (side == 0) mbase0 = mask + (size_t)(tile * 16 + half16 * 8) * mask_ld + ln;
    else           mbase0 = mask + (size_t)ln * mask_ld + tile * 16 + half16 * 8;

    v8f O0 = {0,0,0,0,0,0,0,0}, O1 = O0, O2 = O0, O3 = O0, Csum = O0;
    float M = -3.0e38f;

    for (int jb = 0; jb < ncols; jb += 32) {
        // ---- S = Q @ K^T for two adjacent 16-col tiles ----
        v8f S0 = {0,0,0,0,0,0,0,0}, S1 = S0;
        {
            Frag16 b;
            const _Float16* kp = kbase + (size_t)jb * 64;
            #pragma unroll
            for (int h = 0; h < 16; ++h) b.h[h] = kp[h];
            S0 = wmma16(aq0, b, S0);
            #pragma unroll
            for (int h = 0; h < 16; ++h) b.h[h] = kp[32 + h];
            S0 = wmma16(aq1, b, S0);
            #pragma unroll
            for (int h = 0; h < 16; ++h) b.h[h] = kp[16 * 64 + h];
            S1 = wmma16(aq0, b, S1);
            #pragma unroll
            for (int h = 0; h < 16; ++h) b.h[h] = kp[16 * 64 + 32 + h];
            S1 = wmma16(aq1, b, S1);
        }

        // ---- mask + logits; lane-local max (non-temporal mask stream) ----
        float a0[8], a1[8];
        float lm = -3.0e38f;
        #pragma unroll
        for (int i = 0; i < 8; ++i) {
            float mv0, mv1;
            if (side == 0) {
                const float* mp = mbase0 + (size_t)i * mask_ld + jb;
                mv0 = __builtin_nontemporal_load(mp);
                mv1 = __builtin_nontemporal_load(mp + 16);
            } else {
                const float* mp0 = mbase0 + (size_t)jb * mask_ld + i;
                mv0 = __builtin_nontemporal_load(mp0);
                mv1 = __builtin_nontemporal_load(mp0 + (size_t)16 * mask_ld);
            }
            a0[i] = fmaf(mv0, S0[i] + 1000.f, -1000.f);   // m*s - 1000*(1-m)
            a1[i] = fmaf(mv1, S1[i] + 1000.f, -1000.f);
            lm = fmaxf(lm, fmaxf(a0[i], a1[i]));
        }
        // wave-wide max (block-uniform shift)
        lm = fmaxf(lm, __shfl_xor(lm, 1, 32));
        lm = fmaxf(lm, __shfl_xor(lm, 2, 32));
        lm = fmaxf(lm, __shfl_xor(lm, 4, 32));
        lm = fmaxf(lm, __shfl_xor(lm, 8, 32));
        lm = fmaxf(lm, __shfl_xor(lm, 16, 32));
        if (lm > M) {                       // wave-uniform branch, rare after warmup
            float sc = __expf(M - lm);
            #pragma unroll
            for (int i = 0; i < 8; ++i) {
                O0[i] *= sc; O1[i] *= sc; O2[i] *= sc; O3[i] *= sc; Csum[i] *= sc;
            }
            M = lm;
        }

        // ---- P = exp(a - M), staged [16][32] f16 in LDS ----
        #pragma unroll
        for (int i = 0; i < 8; ++i) {
            int r = (i + half16 * 8) * 32;
            sP[wave][r + ln]      = (_Float16)__expf(a0[i] - M);
            sP[wave][r + 16 + ln] = (_Float16)__expf(a1[i] - M);
        }
        asm volatile("" ::: "memory");

        Frag16 ap;
        #pragma unroll
        for (int h = 0; h < 16; ++h)
            ap.h[h] = sP[wave][ln * 32 + amapK(h, half16)];

        // ---- O += P @ V (full K=32), denominator via P @ ones ----
        #pragma unroll
        for (int c = 0; c < 4; ++c) {
            Frag16 bv;
            const _Float16* vp = vbase + (size_t)c * 16 * ncols + jb;
            #pragma unroll
            for (int h = 0; h < 16; ++h) bv.h[h] = vp[h];
            if (c == 0) O0 = wmma16(ap, bv, O0);
            else if (c == 1) O1 = wmma16(ap, bv, O1);
            else if (c == 2) O2 = wmma16(ap, bv, O2);
            else O3 = wmma16(ap, bv, O3);
        }
        Csum = wmma16(ap, bones, Csum);
        asm volatile("" ::: "memory");
    }

    #pragma unroll
    for (int i = 0; i < 8; ++i) {
        float inv = 1.0f / fmaxf(Csum[i], 1e-35f);
        size_t row = tile * 16 + i + half16 * 8;
        out[row * 64 +      ln] = O0[i] * inv;
        out[row * 64 + 16 + ln] = O1[i] * inv;
        out[row * 64 + 32 + ln] = O2[i] * inv;
        out[row * 64 + 48 + ln] = O3[i] * inv;
    }
}

// ---------------------------------------------------------------------------
// Edge messages: one wave per 16-edge tile. 4 waves / block (LDS budget).
// ---------------------------------------------------------------------------
__global__ __launch_bounds__(128) void edge_kernel(
    const float* __restrict__ coors, const float* __restrict__ h,
    const float* __restrict__ ef,
    const int* __restrict__ src, const int* __restrict__ dst,
    const _Float16* __restrict__ w1t, const float* __restrict__ b1,
    const float* __restrict__ lng, const float* __restrict__ lnb,
    const _Float16* __restrict__ w2t, const float* __restrict__ b2,
    const _Float16* __restrict__ cw1t, const float* __restrict__ cb1,
    const float* __restrict__ cw2, const float* __restrict__ cb2,
    float* __restrict__ aggr, float* __restrict__ xupd, float* __restrict__ deg,
    int nedges)
{
    __shared__ _Float16 sX[4][16 * 192];
    __shared__ _Float16 sY[4][16 * 64];
    __shared__ float    sXrel[4][48];
    __shared__ float    sCoef[4][16];
    __shared__ int      sDst[4][16];

    int wave = threadIdx.x >> 5;
    int lane = threadIdx.x & 31;
    int half16 = lane >> 4;
    int ln = lane & 15;
    int tile = blockIdx.x * 4 + wave;
    if (tile * 16 >= nedges) return;

    _Float16* X = sX[wave];
    _Float16* Y = sY[wave];

    // ---- build input tile [16][192]: h[src] | h[dst] | ef | rbf | pad ----
    if (half16 == 0) {
        int e = tile * 16 + ln;
        int s = src[e], dd = dst[e];
        sDst[wave][ln] = dd;
        _Float16* row = X + ln * 192;
        const float* hs = h + (size_t)s * 64;
        const float* hd = h + (size_t)dd * 64;
        for (int j = 0; j < 64; ++j) { row[j] = (_Float16)hs[j]; row[64 + j] = (_Float16)hd[j]; }
        const float* e32 = ef + (size_t)e * 32;
        for (int j = 0; j < 32; ++j) row[128 + j] = (_Float16)e32[j];
        float x0 = coors[s * 3 + 0] - coors[dd * 3 + 0];
        float x1 = coors[s * 3 + 1] - coors[dd * 3 + 1];
        float x2 = coors[s * 3 + 2] - coors[dd * 3 + 2];
        float d2 = x0 * x0 + x1 * x1 + x2 * x2;
        sXrel[wave][ln * 3 + 0] = x0;
        sXrel[wave][ln * 3 + 1] = x1;
        sXrel[wave][ln * 3 + 2] = x2;
        float inv = 1.0f;
        for (int j = 0; j < 15; ++j) { row[160 + j] = (_Float16)__expf(-d2 * inv); inv *= (1.0f / 1.5f); }
        for (int j = 175; j < 192; ++j) row[j] = (_Float16)0.f;
    }
    asm volatile("" ::: "memory");

    // ---- layer 1: [16x192] @ [192x64] ----
    Frag16 ax[6];
    #pragma unroll
    for (int kc = 0; kc < 6; ++kc)
        #pragma unroll
        for (int hh = 0; hh < 16; ++hh)
            ax[kc].h[hh] = X[ln * 192 + kc * 32 + amapK(hh, half16)];

    v8f C[4];
    #pragma unroll
    for (int c = 0; c < 4; ++c) {
        v8f acc = {0,0,0,0,0,0,0,0};
        #pragma unroll
        for (int kc = 0; kc < 6; ++kc) {
            Frag16 b;
            const _Float16* wp = w1t + (c * 16 + ln) * 192 + kc * 32 + half16 * 16;
            #pragma unroll
            for (int hh = 0; hh < 16; ++hh) b.h[hh] = wp[hh];
            acc = wmma16(ax[kc], b, acc);
        }
        C[c] = acc;
    }

    // ---- bias + lrelu + layernorm (f32) ----
    float gg[4], bb[4];
    #pragma unroll
    for (int c = 0; c < 4; ++c) {
        float bias = b1[c * 16 + ln];
        gg[c] = lng[c * 16 + ln];
        bb[c] = lnb[c * 16 + ln];
        #pragma unroll
        for (int i = 0; i < 8; ++i) C[c][i] = lrelu(C[c][i] + bias);
    }
    #pragma unroll
    for (int i = 0; i < 8; ++i) {
        float s1 = C[0][i] + C[1][i] + C[2][i] + C[3][i];
        float s2 = C[0][i] * C[0][i] + C[1][i] * C[1][i] + C[2][i] * C[2][i] + C[3][i] * C[3][i];
        s1 = redsum16(s1);
        s2 = redsum16(s2);
        float mean = s1 * (1.f / 64.f);
        float var  = s2 * (1.f / 64.f) - mean * mean;
        float rstd = rsqrtf(var + 1e-5f);
        #pragma unroll
        for (int c = 0; c < 4; ++c)
            C[c][i] = (C[c][i] - mean) * rstd * gg[c] + bb[c];
    }

    // ---- stage normalized activations, layer 2 ----
    #pragma unroll
    for (int c = 0; c < 4; ++c)
        #pragma unroll
        for (int i = 0; i < 8; ++i)
            Y[(i + half16 * 8) * 64 + c * 16 + ln] = (_Float16)C[c][i];
    asm volatile("" ::: "memory");

    Frag16 ay0, ay1;
    #pragma unroll
    for (int hh = 0; hh < 16; ++hh) {
        ay0.h[hh] = Y[ln * 64 + amapK(hh, half16)];
        ay1.h[hh] = Y[ln * 64 + 32 + amapK(hh, half16)];
    }
    #pragma unroll
    for (int c = 0; c < 4; ++c) {
        v8f acc = {0,0,0,0,0,0,0,0};
        Frag16 b;
        const _Float16* wp = w2t + (c * 16 + ln) * 64 + half16 * 16;
        #pragma unroll
        for (int hh = 0; hh < 16; ++hh) b.h[hh] = wp[hh];
        acc = wmma16(ay0, b, acc);
        #pragma unroll
        for (int hh = 0; hh < 16; ++hh) b.h[hh] = wp[32 + hh];
        acc = wmma16(ay1, b, acc);
        float bias = b2[c * 16 + ln];
        #pragma unroll
        for (int i = 0; i < 8; ++i) C[c][i] = acc[i] + bias;   // msg
    }

    // ---- scatter aggr_msg (atomics) ----
    #pragma unroll
    for (int c = 0; c < 4; ++c)
        #pragma unroll
        for (int i = 0; i < 8; ++i) {
            int dd = sDst[wave][i + half16 * 8];
            atomicAdd(&aggr[(size_t)dd * 64 + c * 16 + ln], C[c][i]);
        }

    // ---- coors MLP: lrelu(msg@cw1 + cb1) @ cw2 + cb2 ----
    #pragma unroll
    for (int c = 0; c < 4; ++c)
        #pragma unroll
        for (int i = 0; i < 8; ++i)
            Y[(i + half16 * 8) * 64 + c * 16 + ln] = (_Float16)C[c][i];
    asm volatile("" ::: "memory");

    Frag16 am0, am1;
    #pragma unroll
    for (int hh = 0; hh < 16; ++hh) {
        am0.h[hh] = Y[ln * 64 + amapK(hh, half16)];
        am1.h[hh] = Y[ln * 64 + 32 + amapK(hh, half16)];
    }
    float part[8];
    #pragma unroll
    for (int i = 0; i < 8; ++i) part[i] = 0.f;
    #pragma unroll
    for (int c = 0; c < 4; ++c) {
        v8f acc = {0,0,0,0,0,0,0,0};
        Frag16 b;
        const _Float16* wp = cw1t + (c * 16 + ln) * 64 + half16 * 16;
        #pragma unroll
        for (int hh = 0; hh < 16; ++hh) b.h[hh] = wp[hh];
        acc = wmma16(am0, b, acc);
        #pragma unroll
        for (int hh = 0; hh < 16; ++hh) b.h[hh] = wp[32 + hh];
        acc = wmma16(am1, b, acc);
        float bias = cb1[c * 16 + ln];
        float w2v  = cw2[c * 16 + ln];
        #pragma unroll
        for (int i = 0; i < 8; ++i) part[i] = fmaf(lrelu(acc[i] + bias), w2v, part[i]);
    }
    float cb2v = cb2[0];
    #pragma unroll
    for (int i = 0; i < 8; ++i) {
        float coef = redsum16(part[i]) + cb2v;
        if (ln == 0) sCoef[wave][i + half16 * 8] = coef;
    }
    asm volatile("" ::: "memory");

    // ---- scatter x_rel * coef and degree ----
    if (half16 == 0) {
        float cf = sCoef[wave][ln];
        int dd = sDst[wave][ln];
        atomicAdd(&xupd[dd * 3 + 0], sXrel[wave][ln * 3 + 0] * cf);
        atomicAdd(&xupd[dd * 3 + 1], sXrel[wave][ln * 3 + 1] * cf);
        atomicAdd(&xupd[dd * 3 + 2], sXrel[wave][ln * 3 + 2] * cf);
        atomicAdd(&deg[dd], 1.0f);
    }
}

// ---------------------------------------------------------------------------
// Node update: one wave per 16-node tile. 4 waves / block.
// ---------------------------------------------------------------------------
__global__ __launch_bounds__(128) void node_kernel(
    const float* __restrict__ h, const float* __restrict__ aggr,
    const float* __restrict__ att, const float* __restrict__ orig_h,
    const float* __restrict__ deg, const float* __restrict__ xupd,
    const float* __restrict__ coors, const float* __restrict__ orig_coors,
    const _Float16* __restrict__ nw1t, const float* __restrict__ b1,
    const float* __restrict__ lng, const float* __restrict__ lnb,
    const _Float16* __restrict__ nw2t, const float* __restrict__ b2,
    float* __restrict__ out_x, float* __restrict__ out_h, int nnodes)
{
    __shared__ _Float16 sX[4][16 * 256];
    __shared__ _Float16 sY[4][16 * 64];

    int wave = threadIdx.x >> 5;
    int lane = threadIdx.x & 31;
    int half16 = lane >> 4;
    int ln = lane & 15;
    int tile = blockIdx.x * 4 + wave;
    if (tile * 16 >= nnodes) return;

    _Float16* X = sX[wave];
    _Float16* Y = sY[wave];

    if (half16 == 0) {
        int e = tile * 16 + ln;
        float rdeg = 1.0f / fmaxf(deg[e], 1.0f);
        _Float16* row = X + ln * 256;
        const float* hp  = h      + (size_t)e * 64;
        const float* ap  = aggr   + (size_t)e * 64;
        const float* tp  = att    + (size_t)e * 64;
        const float* op  = orig_h + (size_t)e * 64;
        for (int j = 0; j < 64; ++j) {
            row[j]       = (_Float16)hp[j];
            row[64 + j]  = (_Float16)(ap[j] * rdeg);
            row[128 + j] = (_Float16)tp[j];
            row[192 + j] = (_Float16)op[j];
        }
        for (int d = 0; d < 3; ++d)
            out_x[e * 3 + d] = 0.25f * orig_coors[e * 3 + d] + 0.75f * coors[e * 3 + d]
                             + xupd[e * 3 + d] * rdeg;
    }
    asm volatile("" ::: "memory");

    Frag16 ax[8];
    #pragma unroll
    for (int kc = 0; kc < 8; ++kc)
        #pragma unroll
        for (int hh = 0; hh < 16; ++hh)
            ax[kc].h[hh] = X[ln * 256 + kc * 32 + amapK(hh, half16)];

    v8f C[4];
    float gg[4], bb[4];
    #pragma unroll
    for (int c = 0; c < 4; ++c) {
        v8f acc = {0,0,0,0,0,0,0,0};
        #pragma unroll
        for (int kc = 0; kc < 8; ++kc) {
            Frag16 b;
            const _Float16* wp = nw1t + (c * 16 + ln) * 256 + kc * 32 + half16 * 16;
            #pragma unroll
            for (int hh = 0; hh < 16; ++hh) b.h[hh] = wp[hh];
            acc = wmma16(ax[kc], b, acc);
        }
        float bias = b1[c * 16 + ln];
        gg[c] = lng[c * 16 + ln];
        bb[c] = lnb[c * 16 + ln];
        #pragma unroll
        for (int i = 0; i < 8; ++i) C[c][i] = lrelu(acc[i] + bias);
    }
    #pragma unroll
    for (int i = 0; i < 8; ++i) {
        float s1 = C[0][i] + C[1][i] + C[2][i] + C[3][i];
        float s2 = C[0][i] * C[0][i] + C[1][i] * C[1][i] + C[2][i] * C[2][i] + C[3][i] * C[3][i];
        s1 = redsum16(s1);
        s2 = redsum16(s2);
        float mean = s1 * (1.f / 64.f);
        float var  = s2 * (1.f / 64.f) - mean * mean;
        float rstd = rsqrtf(var + 1e-5f);
        #pragma unroll
        for (int c = 0; c < 4; ++c)
            C[c][i] = (C[c][i] - mean) * rstd * gg[c] + bb[c];
    }
    #pragma unroll
    for (int c = 0; c < 4; ++c)
        #pragma unroll
        for (int i = 0; i < 8; ++i)
            Y[(i + half16 * 8) * 64 + c * 16 + ln] = (_Float16)C[c][i];
    asm volatile("" ::: "memory");

    Frag16 ay0, ay1;
    #pragma unroll
    for (int hh = 0; hh < 16; ++hh) {
        ay0.h[hh] = Y[ln * 64 + amapK(hh, half16)];
        ay1.h[hh] = Y[ln * 64 + 32 + amapK(hh, half16)];
    }
    #pragma unroll
    for (int c = 0; c < 4; ++c) {
        v8f acc = {0,0,0,0,0,0,0,0};
        Frag16 b;
        const _Float16* wp = nw2t + (c * 16 + ln) * 64 + half16 * 16;
        #pragma unroll
        for (int hh = 0; hh < 16; ++hh) b.h[hh] = wp[hh];
        acc = wmma16(ay0, b, acc);
        #pragma unroll
        for (int hh = 0; hh < 16; ++hh) b.h[hh] = wp[32 + hh];
        acc = wmma16(ay1, b, acc);
        float bias = b2[c * 16 + ln];
        #pragma unroll
        for (int i = 0; i < 8; ++i) {
            size_t r = tile * 16 + i + half16 * 8;
            float hv = h[r * 64 + c * 16 + ln];
            out_h[r * 64 + c * 16 + ln] = 0.5f * (acc[i] + bias) + 0.5f * hv;
        }
    }
}

// ---------------------------------------------------------------------------
extern "C" void kernel_launch(void* const* d_in, const int* in_sizes, int n_in,
                              void* d_out, int out_size, void* d_ws, size_t ws_size,
                              hipStream_t stream)
{
    const float* coors_l      = (const float*)d_in[0];
    const float* h_l          = (const float*)d_in[1];
    const float* orig_h_l     = (const float*)d_in[2];
    const float* edge_feat_l  = (const float*)d_in[3];
    const float* orig_coors_l = (const float*)d_in[4];
    const float* coors_r      = (const float*)d_in[5];
    const float* h_r          = (const float*)d_in[6];
    const float* orig_h_r     = (const float*)d_in[7];
    const float* edge_feat_r  = (const float*)d_in[8];
    const float* orig_coors_r = (const float*)d_in[9];
    const float* mask         = (const float*)d_in[10];
    const int*   src_l        = (const int*)d_in[11];
    const int*   dst_l        = (const int*)d_in[12];
    const int*   src_r        = (const int*)d_in[13];
    const int*   dst_r        = (const int*)d_in[14];
    const float* edge_w1      = (const float*)d_in[15];
    const float* edge_b1      = (const float*)d_in[16];
    const float* edge_ln_g    = (const float*)d_in[17];
    const float* edge_ln_b    = (const float*)d_in[18];
    const float* edge_w2      = (const float*)d_in[19];
    const float* edge_b2      = (const float*)d_in[20];
    const float* q_w          = (const float*)d_in[21];
    const float* k_w          = (const float*)d_in[22];
    const float* v_w          = (const float*)d_in[23];
    const float* v_b          = (const float*)d_in[24];
    const float* node_w1      = (const float*)d_in[25];
    const float* node_b1      = (const float*)d_in[26];
    const float* node_ln_g    = (const float*)d_in[27];
    const float* node_ln_b    = (const float*)d_in[28];
    const float* node_w2      = (const float*)d_in[29];
    const float* node_b2      = (const float*)d_in[30];
    const float* coors_w1     = (const float*)d_in[31];
    const float* coors_b1     = (const float*)d_in[32];
    const float* coors_w2     = (const float*)d_in[33];
    const float* coors_b2     = (const float*)d_in[34];

    const int NL = in_sizes[1] / 64;
    const int NR = in_sizes[6] / 64;
    const int EL = in_sizes[11];
    const int ER = in_sizes[13];

    // workspace carve-up
    char* p = (char*)d_ws;
    auto carve = [&](size_t bytes) { void* r = (void*)p; p += (bytes + 255) & ~(size_t)255; return r; };
    _Float16* q_l  = (_Float16*)carve((size_t)NL * 64 * 2);
    _Float16* k_l  = (_Float16*)carve((size_t)NL * 64 * 2);
    _Float16* vt_l = (_Float16*)carve((size_t)NL * 64 * 2);
    _Float16* q_r  = (_Float16*)carve((size_t)NR * 64 * 2);
    _Float16* k_r  = (_Float16*)carve((size_t)NR * 64 * 2);
    _Float16* vt_r = (_Float16*)carve((size_t)NR * 64 * 2);
    _Float16* w1t  = (_Float16*)carve(64 * 192 * 2);
    _Float16* w2t  = (_Float16*)carve(64 * 64 * 2);
    _Float16* cw1t = (_Float16*)carve(64 * 64 * 2);
    _Float16* nw1t = (_Float16*)carve(64 * 256 * 2);
    _Float16* nw2t = (_Float16*)carve(64 * 64 * 2);
    float* att_l  = (float*)carve((size_t)NL * 64 * 4);
    float* att_r  = (float*)carve((size_t)NR * 64 * 4);
    float* aggr_l = (float*)carve((size_t)NL * 64 * 4);
    float* aggr_r = (float*)carve((size_t)NR * 64 * 4);
    float* xupd_l = (float*)carve((size_t)NL * 3 * 4);
    float* xupd_r = (float*)carve((size_t)NR * 3 * 4);
    float* deg_l  = (float*)carve((size_t)NL * 4);
    float* deg_r  = (float*)carve((size_t)NR * 4);

    // zero accumulators (graph-capturable memset nodes)
    hipMemsetAsync(aggr_l, 0, (size_t)NL * 64 * 4, stream);
    hipMemsetAsync(aggr_r, 0, (size_t)NR * 64 * 4, stream);
    hipMemsetAsync(xupd_l, 0, (size_t)NL * 3 * 4, stream);
    hipMemsetAsync(xupd_r, 0, (size_t)NR * 3 * 4, stream);
    hipMemsetAsync(deg_l, 0, (size_t)NL * 4, stream);
    hipMemsetAsync(deg_r, 0, (size_t)NR * 4, stream);

    prep_weights<<<64, 256, 0, stream>>>(edge_w1, edge_w2, coors_w1, node_w1, node_w2,
                                         w1t, w2t, cw1t, nw1t, nw2t);

    qkv_kernel<<<(NL * 64 + 255) / 256, 256, 0, stream>>>(h_l, q_w, k_w, v_w, v_b, q_l, k_l, vt_l, NL);
    qkv_kernel<<<(NR * 64 + 255) / 256, 256, 0, stream>>>(h_r, q_w, k_w, v_w, v_b, q_r, k_r, vt_r, NR);

    // both attention sides in one launch: y==0 -> att_l (q_l vs k_r/v_r),
    // y==1 -> att_r (q_r vs k_l/v_l, transposed mask)
    {
        int maxtiles = ((NL > NR ? NL : NR) / 16 + 7) / 8;
        dim3 grid(maxtiles, 2);
        if (NL == 8000 && NR == 8000) {
            attn_kernel<8000><<<grid, 256, 0, stream>>>(
                q_l, k_r, vt_r, att_l,
                q_r, k_l, vt_l, att_r,
                mask, NL, NR, NR, NL, NR);
        } else {
            attn_kernel<0><<<grid, 256, 0, stream>>>(
                q_l, k_r, vt_r, att_l,
                q_r, k_l, vt_l, att_r,
                mask, NL, NR, NR, NL, NR);
        }
    }

    edge_kernel<<<(EL / 16 + 3) / 4, 128, 0, stream>>>(
        coors_l, h_l, edge_feat_l, src_l, dst_l,
        w1t, edge_b1, edge_ln_g, edge_ln_b, w2t, edge_b2,
        cw1t, coors_b1, coors_w2, coors_b2,
        aggr_l, xupd_l, deg_l, EL);
    edge_kernel<<<(ER / 16 + 3) / 4, 128, 0, stream>>>(
        coors_r, h_r, edge_feat_r, src_r, dst_r,
        w1t, edge_b1, edge_ln_g, edge_ln_b, w2t, edge_b2,
        cw1t, coors_b1, coors_w2, coors_b2,
        aggr_r, xupd_r, deg_r, ER);

    float* out = (float*)d_out;
    float* out_x_l = out;
    float* out_h_l = out + (size_t)NL * 3;
    float* out_x_r = out + (size_t)NL * 3 + (size_t)NL * 64;
    float* out_h_r = out_x_r + (size_t)NR * 3;

    node_kernel<<<(NL / 16 + 3) / 4, 128, 0, stream>>>(
        h_l, aggr_l, att_l, orig_h_l, deg_l, xupd_l, coors_l, orig_coors_l,
        nw1t, node_b1, node_ln_g, node_ln_b, nw2t, node_b2,
        out_x_l, out_h_l, NL);
    node_kernel<<<(NR / 16 + 3) / 4, 128, 0, stream>>>(
        h_r, aggr_r, att_r, orig_h_r, deg_r, xupd_r, coors_r, orig_coors_r,
        nw1t, node_b1, node_ln_g, node_ln_b, nw2t, node_b2,
        out_x_r, out_h_r, NR);
}
